// EdgeClassifierGNN_65704409694580
// MI455X (gfx1250) — matrix operand order repaired and checked by reading, not compiled, not run on hardware
//
#include <hip/hip_runtime.h>

typedef __attribute__((ext_vector_type(16))) _Float16 v16h;
typedef __attribute__((ext_vector_type(8)))  _Float16 v8h;
typedef __attribute__((ext_vector_type(8)))  float    v8f;

#define WPB 4  // waves per block (wave32 -> blockDim 128)

// ---------------------------------------------------------------------------
// CDNA5 async global->LDS staging (ASYNCcnt path, ISA 15.18.3 / 08_async_tensor)
// ---------------------------------------------------------------------------
__device__ __forceinline__ void async_g2l_b128(unsigned lds_off, const void* g)
{
    asm volatile("global_load_async_to_lds_b128 %0, %1, off"
                 :: "v"(lds_off), "v"(g) : "memory");
}
__device__ __forceinline__ void wait_async0()
{
    asm volatile("s_wait_asynccnt 0x0" ::: "memory");
}

// Stage nchunks x 16B of pre-swizzled B fragments into LDS (whole block).
__device__ __forceinline__ void stage_B(const _Float16* __restrict__ Bsw,
                                        unsigned shbase, int nchunks)
{
    for (int i = threadIdx.x; i < nchunks; i += blockDim.x)
        async_g2l_b128(shbase + (unsigned)i * 16u, (const char*)Bsw + (size_t)i * 16);
    wait_async0();      // my async ops done
    __syncthreads();    // everyone's staging visible
}

// ---------------------------------------------------------------------------
// Weight prep: pack [Wl | Wr] into per-wave WMMA B fragments:
//   frag f = kt*8+nt; lane<16:  K = kt*32 + j (j=0..15), col = nt*16+lane
//                     lane>=16: K = kt*32 + 16 + j,      col = nt*16+lane-16
// 32 lanes x 16 halfs, contiguous per lane (one v16h per lane per fragment).
// ---------------------------------------------------------------------------
__global__ void k_prep_w(const float* __restrict__ Wl, const float* __restrict__ Wr,
                         int kin, int kpad, int ldw, int ktiles,
                         _Float16* __restrict__ Bsw)
{
    int t = blockIdx.x * blockDim.x + threadIdx.x;      // one per (kt, nt, lane)
    int total = ktiles * 8 * 32;
    if (t >= total) return;
    int lane = t & 31;
    int nt   = (t >> 5) & 7;
    int kt   = t >> 8;
    int n     = nt * 16 + (lane & 15);
    int kbase = kt * 32 + ((lane >> 4) * 16);
    _Float16* o = Bsw + (size_t)t * 16;
    #pragma unroll
    for (int j = 0; j < 16; ++j) {
        int k = kbase + j;
        float v = 0.0f;
        if (k < kpad) {
            if (k < kin) v = Wl[(size_t)n * ldw + k];
        } else {
            int kk = k - kpad;
            if (Wr != nullptr && kk < kin) v = Wr[(size_t)n * ldw + kk];
        }
        o[j] = (_Float16)v;
    }
}

// ---------------------------------------------------------------------------
// Stage-1 scatter: per edge, rebuild h0[src] on the fly and atomically
// accumulate into agg[dst] (48-wide padded) + degree.
// ---------------------------------------------------------------------------
__global__ void k_scatter1(const float* __restrict__ x, const float* __restrict__ emb,
                           const int* __restrict__ srcI, const int* __restrict__ dstI,
                           float* __restrict__ agg, float* __restrict__ deg, int E)
{
    int e = blockIdx.x * blockDim.x + threadIdx.x;
    if (e >= E) return;
    int s = srcI[e], d = dstI[e];
    const float* xr = x + (size_t)s * 10;
    float* ar = agg + (size_t)d * 48;
    atomicAdd(ar + 0, xr[0]);
    #pragma unroll
    for (int j = 0; j < 8; ++j) atomicAdd(ar + 1 + j, xr[2 + j]);
    int pid = (int)xr[1];
    const float* ev = emb + (size_t)pid * 32;
    for (int j = 0; j < 32; ++j) atomicAdd(ar + 9 + j, ev[j]);
    atomicAdd(deg + d, 1.0f);
}

// A1 row (96 halfs) = [mean(41) pad48 | h0(41) pad48]   (Wl multiplies mean)
__global__ void k_finalize1(const float* __restrict__ x, const float* __restrict__ emb,
                            const float* __restrict__ agg, const float* __restrict__ deg,
                            _Float16* __restrict__ A1, int N)
{
    int i = blockIdx.x * blockDim.x + threadIdx.x;
    if (i >= N) return;
    float invd = 1.0f / fmaxf(deg[i], 1.0f);
    const float* ar = agg + (size_t)i * 48;
    _Float16* o = A1 + (size_t)i * 96;
    for (int j = 0; j < 41; ++j) o[j] = (_Float16)(ar[j] * invd);
    for (int j = 41; j < 48; ++j) o[j] = (_Float16)0.0f;
    const float* xr = x + (size_t)i * 10;
    o[48] = (_Float16)xr[0];
    #pragma unroll
    for (int j = 0; j < 8; ++j) o[49 + j] = (_Float16)xr[2 + j];
    int pid = (int)xr[1];
    const float* ev = emb + (size_t)pid * 32;
    for (int j = 0; j < 32; ++j) o[57 + j] = (_Float16)ev[j];
    for (int j = 89; j < 96; ++j) o[j] = (_Float16)0.0f;
}

// Stage-2 scatter: 4 threads per edge, 32 feats each.
__global__ void k_scatter2(const int* __restrict__ srcI, const int* __restrict__ dstI,
                           const float* __restrict__ h1, float* __restrict__ agg, int E)
{
    int t = blockIdx.x * blockDim.x + threadIdx.x;
    if (t >= E * 4) return;
    int e = t >> 2, c = (t & 3) * 32;
    int s = srcI[e], d = dstI[e];
    const float* hv = h1 + (size_t)s * 128 + c;
    float* ar = agg + (size_t)d * 128 + c;
    #pragma unroll
    for (int j = 0; j < 32; ++j) atomicAdd(ar + j, hv[j]);
}

// A2 row (256 halfs) = [mean1(128) | h1(128)]
__global__ void k_finalize2(const float* __restrict__ h1, const float* __restrict__ agg,
                            const float* __restrict__ deg, _Float16* __restrict__ A2, int N)
{
    int i = blockIdx.x * blockDim.x + threadIdx.x;
    if (i >= N) return;
    float invd = 1.0f / fmaxf(deg[i], 1.0f);
    _Float16* o = A2 + (size_t)i * 256;
    const float* ar = agg + (size_t)i * 128;
    const float* hr = h1 + (size_t)i * 128;
    for (int j = 0; j < 128; ++j) o[j]       = (_Float16)(ar[j] * invd);
    for (int j = 0; j < 128; ++j) o[128 + j] = (_Float16)hr[j];
}

// ---------------------------------------------------------------------------
// Node-layer GEMM: B panel async-staged to LDS once per block; one wave does
// 32 rows x 128 cols (2 M-tiles, 16 accumulators), f16 WMMA / f32 accum.
// Epilogue (LDS tile aliases the dead B panel): +bias, row L2-norm, ReLU.
// ---------------------------------------------------------------------------
template<int KT>
__global__ __launch_bounds__(WPB * 32)
void k_gemm_nodes(const _Float16* __restrict__ A, int lda,
                  const _Float16* __restrict__ Bsw, const float* __restrict__ bias,
                  int M, float* __restrict__ outF, _Float16* __restrict__ outH)
{
    __shared__ __align__(128) unsigned char shraw[64 * 1024];
    _Float16* shB  = (_Float16*)shraw;   // KT*8 KB used during main loop
    float*    tile = (float*)shraw;      // WPB*32*128 f32 = 64KB, after B is dead

    stage_B(Bsw, (unsigned)(size_t)(void*)shraw, KT * 8 * 64);

    int wave = threadIdx.x >> 5;
    int lane = threadIdx.x & 31;
    int row0 = (blockIdx.x * WPB + wave) * 32;
    bool active = row0 < M;                        // wave-uniform

    v8f acc0[8], acc1[8];
    v8f zero = {0.f, 0.f, 0.f, 0.f, 0.f, 0.f, 0.f, 0.f};
    #pragma unroll
    for (int i = 0; i < 8; ++i) { acc0[i] = zero; acc1[i] = zero; }

    if (active) {
        int r0 = row0 + (lane & 15);        if (r0 >= M) r0 = M - 1;
        int r1 = row0 + 16 + (lane & 15);   if (r1 >= M) r1 = M - 1;
        const _Float16* a0p = A + (size_t)r0 * lda;
        const _Float16* a1p = A + (size_t)r1 * lda;
        int sel8 = (lane >> 4) * 16;        // hi lanes hold K+16..K+31
        #pragma unroll
        for (int kt = 0; kt < KT; ++kt) {
            v8h lo0 = *(const v8h*)(a0p + kt * 32 + sel8);
            v8h hi0 = *(const v8h*)(a0p + kt * 32 + 8 + sel8);
            v8h lo1 = *(const v8h*)(a1p + kt * 32 + sel8);
            v8h hi1 = *(const v8h*)(a1p + kt * 32 + 8 + sel8);
            v16h a0 = __builtin_shufflevector(lo0, hi0, 0,1,2,3,4,5,6,7,8,9,10,11,12,13,14,15);
            v16h a1 = __builtin_shufflevector(lo1, hi1, 0,1,2,3,4,5,6,7,8,9,10,11,12,13,14,15);
            const _Float16* bp = shB + ((size_t)(kt * 8) * 32 + lane) * 16;
            #pragma unroll
            for (int nt = 0; nt < 8; ++nt) {
                v16h b = *(const v16h*)(bp + (size_t)nt * 512);
                acc0[nt] = __builtin_amdgcn_wmma_f32_16x16x32_f16(
                    false, a0, false, b, (short)0, acc0[nt], false, false);
                acc1[nt] = __builtin_amdgcn_wmma_f32_16x16x32_f16(
                    false, a1, false, b, (short)0, acc1[nt], false, false);
            }
        }
    }
    __syncthreads();   // all waves done reading shB; tile may alias it now
    if (active) {
        float* tw = tile + (size_t)wave * 32 * 128;
        // C layout: lane<16 -> rows v, col=lane ; lane>=16 -> rows v+8, col=lane-16
        int col0 = lane & 15;
        int half8 = (lane >> 4) * 8;
        #pragma unroll
        for (int nt = 0; nt < 8; ++nt) {
            int c = nt * 16 + col0;
            float bv = bias[c];
            #pragma unroll
            for (int v = 0; v < 8; ++v) {
                tw[(size_t)(half8 + v) * 128 + c]      = acc0[nt][v] + bv;
                tw[(size_t)(16 + half8 + v) * 128 + c] = acc1[nt][v] + bv;
            }
        }
        __builtin_amdgcn_wave_barrier();   // keep DS write->read order (same wave)
        // lane L owns row L's L2 norm
        float s = 0.0f;
        const float* rp = tw + (size_t)lane * 128;
        for (int c = 0; c < 128; ++c) { float t = rp[c]; s += t * t; }
        float rninv = 1.0f / fmaxf(sqrtf(s), 1e-12f);
        for (int i = lane; i < 32 * 128; i += 32) {
            int rr = i >> 7, cc = i & 127;          // rr wave-uniform per iter
            float rn = __shfl(rninv, rr, 32);
            int grow = row0 + rr;
            if (grow < M) {
                float v = fmaxf(tw[(size_t)rr * 128 + cc] * rn, 0.0f);
                if (outF) outF[(size_t)grow * 128 + cc] = v;
                if (outH) outH[(size_t)grow * 128 + cc] = (_Float16)v;
            }
        }
    }
}

// ---------------------------------------------------------------------------
// Edge MLP: A row e = [h2[src[e]] | h2[dst[e]]] gathered in f16 (L2-resident);
// K=256 -> 128 hidden (WMMA) -> ReLU -> 128->2 head -> d_out.  B from LDS.
// ---------------------------------------------------------------------------
__global__ __launch_bounds__(WPB * 32)
void k_gemm_edges(const _Float16* __restrict__ H2,
                  const int* __restrict__ srcI, const int* __restrict__ dstI,
                  const _Float16* __restrict__ Bsw, const float* __restrict__ b1,
                  const float* __restrict__ W2, const float* __restrict__ b2,
                  int E, float* __restrict__ out)
{
    __shared__ __align__(128) unsigned char shraw[64 * 1024];
    _Float16* shB  = (_Float16*)shraw;
    float*    tile = (float*)shraw;

    stage_B(Bsw, (unsigned)(size_t)(void*)shraw, 8 * 8 * 64);

    int wave = threadIdx.x >> 5;
    int lane = threadIdx.x & 31;
    int row0 = (blockIdx.x * WPB + wave) * 32;
    bool active = row0 < E;                        // wave-uniform

    v8f acc0[8], acc1[8];
    v8f zero = {0.f, 0.f, 0.f, 0.f, 0.f, 0.f, 0.f, 0.f};
    #pragma unroll
    for (int i = 0; i < 8; ++i) { acc0[i] = zero; acc1[i] = zero; }

    if (active) {
        int e0 = row0 + (lane & 15);        if (e0 >= E) e0 = E - 1;
        int e1 = row0 + 16 + (lane & 15);   if (e1 >= E) e1 = E - 1;
        const _Float16* s0 = H2 + (size_t)srcI[e0] * 128;
        const _Float16* d0 = H2 + (size_t)dstI[e0] * 128;
        const _Float16* s1 = H2 + (size_t)srcI[e1] * 128;
        const _Float16* d1 = H2 + (size_t)dstI[e1] * 128;
        int sel8 = (lane >> 4) * 16;
        #pragma unroll
        for (int kt = 0; kt < 8; ++kt) {
            const _Float16* p0 = (kt < 4 ? s0 + kt * 32 : d0 + (kt - 4) * 32) + sel8;
            const _Float16* p1 = (kt < 4 ? s1 + kt * 32 : d1 + (kt - 4) * 32) + sel8;
            v8h lo0 = *(const v8h*)(p0);
            v8h hi0 = *(const v8h*)(p0 + 8);
            v8h lo1 = *(const v8h*)(p1);
            v8h hi1 = *(const v8h*)(p1 + 8);
            v16h a0 = __builtin_shufflevector(lo0, hi0, 0,1,2,3,4,5,6,7,8,9,10,11,12,13,14,15);
            v16h a1 = __builtin_shufflevector(lo1, hi1, 0,1,2,3,4,5,6,7,8,9,10,11,12,13,14,15);
            const _Float16* bp = shB + ((size_t)(kt * 8) * 32 + lane) * 16;
            #pragma unroll
            for (int nt = 0; nt < 8; ++nt) {
                v16h b = *(const v16h*)(bp + (size_t)nt * 512);
                acc0[nt] = __builtin_amdgcn_wmma_f32_16x16x32_f16(
                    false, a0, false, b, (short)0, acc0[nt], false, false);
                acc1[nt] = __builtin_amdgcn_wmma_f32_16x16x32_f16(
                    false, a1, false, b, (short)0, acc1[nt], false, false);
            }
        }
    }
    __syncthreads();   // shB dead; tile aliases it
    if (active) {
        float* tw = tile + (size_t)wave * 32 * 128;
        int col0 = lane & 15;
        int half8 = (lane >> 4) * 8;
        #pragma unroll
        for (int nt = 0; nt < 8; ++nt) {
            int c = nt * 16 + col0;
            float bv = b1[c];
            #pragma unroll
            for (int v = 0; v < 8; ++v) {
                tw[(size_t)(half8 + v) * 128 + c]      = fmaxf(acc0[nt][v] + bv, 0.0f);
                tw[(size_t)(16 + half8 + v) * 128 + c] = fmaxf(acc1[nt][v] + bv, 0.0f);
            }
        }
        __builtin_amdgcn_wave_barrier();
        // head: 32 rows x 2 outputs; each lane does 2 dots of length 128
        #pragma unroll
        for (int sub = 0; sub < 2; ++sub) {
            int rr = sub * 16 + (lane & 15);
            int c  = lane >> 4;
            int e  = row0 + rr;
            if (e < E) {
                float sacc = b2[c];
                const float* w = W2 + (size_t)c * 128;
                const float* rowp = tw + (size_t)rr * 128;
                for (int k = 0; k < 128; ++k) sacc += rowp[k] * w[k];
                out[(size_t)e * 2 + c] = sacc;
            }
        }
    }
}

// ---------------------------------------------------------------------------
extern "C" void kernel_launch(void* const* d_in, const int* in_sizes, int n_in,
                              void* d_out, int out_size, void* d_ws, size_t ws_size,
                              hipStream_t stream)
{
    const float* x    = (const float*)d_in[0];
    const int*   ei   = (const int*)d_in[1];
    const float* emb  = (const float*)d_in[2];
    const float* Wl1  = (const float*)d_in[3];
    const float* bl1  = (const float*)d_in[4];
    const float* Wr1  = (const float*)d_in[5];
    const float* Wl2  = (const float*)d_in[6];
    const float* bl2  = (const float*)d_in[7];
    const float* Wr2  = (const float*)d_in[8];
    const float* W1m  = (const float*)d_in[9];
    const float* b1   = (const float*)d_in[10];
    const float* W2   = (const float*)d_in[11];
    const float* b2   = (const float*)d_in[12];

    const int N = in_sizes[0] / 10;
    const int E = in_sizes[1] / 2;
    const int* srcI = ei;
    const int* dstI = ei + E;

    // Workspace carve (aliased across phases, ~180 MB total):
    size_t off = 0;
    auto carve = [&](size_t bytes) -> void* {
        void* p = (char*)d_ws + off;
        off += (bytes + 255) & ~(size_t)255;
        return p;
    };
    float*     aggBuf = (float*)carve((size_t)N * 128 * sizeof(float));       // agg0(48w) then agg2(128w)
    _Float16*  Abuf   = (_Float16*)carve((size_t)N * 256 * sizeof(_Float16)); // A1(96w) then A2(256w)
    float*     h1     = (float*)carve((size_t)N * 128 * sizeof(float));
    _Float16*  h2     = (_Float16*)carve((size_t)N * 128 * sizeof(_Float16));
    float*     deg    = (float*)carve((size_t)N * sizeof(float));
    _Float16*  Bsw1   = (_Float16*)carve((size_t)3 * 8 * 32 * 16 * sizeof(_Float16));
    _Float16*  Bsw2   = (_Float16*)carve((size_t)8 * 8 * 32 * 16 * sizeof(_Float16));
    _Float16*  Bsw3   = (_Float16*)carve((size_t)8 * 8 * 32 * 16 * sizeof(_Float16));
    (void)ws_size; (void)n_in; (void)out_size;

    const int TB = 256;
    dim3 blk(TB);
    int gN = (N + WPB * 32 - 1) / (WPB * 32);   // 32 rows per wave now
    int gE = (E + WPB * 32 - 1) / (WPB * 32);

    // Phase 0: zero accumulators, swizzle weights into WMMA fragment layout
    hipMemsetAsync(aggBuf, 0, (size_t)N * 48 * sizeof(float), stream);
    hipMemsetAsync(deg,    0, (size_t)N * sizeof(float), stream);
    k_prep_w<<<3, blk, 0, stream>>>(Wl1, Wr1, 41, 48, 41, 3, Bsw1);
    k_prep_w<<<8, blk, 0, stream>>>(Wl2, Wr2, 128, 128, 128, 8, Bsw2);
    k_prep_w<<<8, blk, 0, stream>>>(W1m, nullptr, 256, 256, 256, 8, Bsw3);

    // Phase 1: SAGE layer 1
    k_scatter1<<<(E + TB - 1) / TB, blk, 0, stream>>>(x, emb, srcI, dstI, aggBuf, deg, E);
    k_finalize1<<<(N + TB - 1) / TB, blk, 0, stream>>>(x, emb, aggBuf, deg, Abuf, N);
    k_gemm_nodes<3><<<gN, dim3(WPB * 32), 0, stream>>>(Abuf, 96, Bsw1, bl1, N, h1, nullptr);

    // Phase 2: SAGE layer 2
    hipMemsetAsync(aggBuf, 0, (size_t)N * 128 * sizeof(float), stream);
    k_scatter2<<<(E * 4 + TB - 1) / TB, blk, 0, stream>>>(srcI, dstI, h1, aggBuf, E);
    k_finalize2<<<(N + TB - 1) / TB, blk, 0, stream>>>(h1, aggBuf, deg, Abuf, N);
    k_gemm_nodes<8><<<gN, dim3(WPB * 32), 0, stream>>>(Abuf, 256, Bsw2, bl2, N, nullptr, h2);

    // Phase 3: edge MLP (f16 gathers of L2-resident h2 + LDS-fed WMMA + head)
    k_gemm_edges<<<gE, dim3(WPB * 32), 0, stream>>>(h2, srcI, dstI, Bsw3, b1, W2, b2,
                                                    E, (float*)d_out);
}
